// Superattention_conv_opt_79903571575313
// MI455X (gfx1250) — matrix operand order repaired and checked by reading, not compiled, not run on hardware
//
#include <hip/hip_runtime.h>
#include <hip/hip_bf16.h>

// Problem constants (match reference)
constexpr int BB = 2;      // batch
constexpr int C  = 256;    // channels
constexpr int N  = 4096;   // target length
constexpr int M  = 4096;   // ref length

// GEMM tiling
constexpr int TILE  = 128; // block tile (rows x cols)
constexpr int KSTEP = 32;  // K per WMMA step (f16 16x16x32)
constexpr int LROW  = 40;  // padded LDS row stride in f16 elems (80B, bank-conflict free)

typedef __attribute__((ext_vector_type(16))) _Float16 v16h;
typedef __attribute__((ext_vector_type(8)))  float    v8f;

union Frag { int4 q[2]; v16h v; };

// ---------------------------------------------------------------------------
// gfx1250 async global->LDS staging (ASYNCcnt-tracked DMA path), with a
// verified-sync fallback if the toolchain lacks the builtins.
// Probe result: param0 is 'int __vector(4) __device__ *' (global AS first,
// native vector pointee) — matching the classic global_load_lds convention.
// ---------------------------------------------------------------------------
#if __has_builtin(__builtin_amdgcn_global_load_async_to_lds_b128) && \
    __has_builtin(__builtin_amdgcn_s_wait_asynccnt)
#define USE_ASYNC_LDS 1
#else
#define USE_ASYNC_LDS 0
#endif

typedef int avec4i __attribute__((vector_size(16)));

__device__ __forceinline__ void copy16_to_lds(void* lds, const void* gbl)
{
#if USE_ASYNC_LDS
    // Pass real pointers (generic) cast to the builtin's vector pointee type;
    // clang applies the implicit generic->AS1/AS3 argument conversions.
    __builtin_amdgcn_global_load_async_to_lds_b128(
        (avec4i*)gbl, (avec4i*)lds, 0, 0);
#else
    *(int4*)lds = *(const int4*)gbl;
#endif
}

__device__ __forceinline__ void staging_commit()
{
#if USE_ASYNC_LDS
    __builtin_amdgcn_s_wait_asynccnt(0);   // s_wait_asynccnt 0
#endif
}

// ---------------------------------------------------------------------------
// Kernel 1: 1x1 conv projection.  out = W (CxC) * x (CxL) + bias, cast to f16.
// transposed==0 : out[b][pos][c]  (K-contiguous layout for WMMA A/B of sim GEMM)
// transposed==1 : out[b][c][pos]  (natural layout, used as B-matrix of P*V^T)
// ---------------------------------------------------------------------------
__global__ __launch_bounds__(256) void proj_kernel(
    const float* __restrict__ x, const float* __restrict__ w,
    const float* __restrict__ bias, _Float16* __restrict__ out, int transposed)
{
    const int tid = threadIdx.x;
    const int b   = blockIdx.z;
    int c, pos;
    if (!transposed) {        // consecutive threads -> consecutive c (coalesced store)
        c   = blockIdx.y * 4  + (tid & 3);
        pos = blockIdx.x * 64 + (tid >> 2);
    } else {                  // consecutive threads -> consecutive pos (coalesced store)
        pos = blockIdx.x * 64 + (tid & 63);
        c   = blockIdx.y * 4  + (tid >> 6);
    }
    const float* xb = x + (size_t)b * C * N;
    const float* wr = w + (size_t)c * C;
    float acc = bias[c];
#pragma unroll 8
    for (int k = 0; k < C; ++k)
        acc += wr[k] * xb[(size_t)k * N + pos];

    size_t idx = transposed ? ((size_t)b * C + c) * (size_t)M + pos
                            : ((size_t)b * N + pos) * (size_t)C + c;
    out[idx] = (_Float16)acc;
}

// ---------------------------------------------------------------------------
// Kernel 2: sim = Q^T K (f16 WMMA, f32 accum) with ==0 -> -1e5 mask fused.
// Q: [b][n][C] f16, K: [b][m][C] f16.  128x128 tile / block, 8 waves.
// Wave w owns columns m = w*16..+15, loops 8 row subtiles.
// ---------------------------------------------------------------------------
__global__ __launch_bounds__(256) void sim_kernel(
    const _Float16* __restrict__ Q, const _Float16* __restrict__ K,
    float* __restrict__ sim)
{
    __shared__ _Float16 Qs[TILE * LROW];
    __shared__ _Float16 Ks[TILE * LROW];

    const int tid = threadIdx.x;
    const int b   = blockIdx.z;
    const int n0  = blockIdx.y * TILE;
    const int m0  = blockIdx.x * TILE;

    const _Float16* Qg = Q + ((size_t)b * N + n0) * C;
    const _Float16* Kg = K + ((size_t)b * M + m0) * C;

    const int w = tid >> 5, lane = tid & 31, g = lane >> 4, r = lane & 15;

    const v8f zero = {0.f,0.f,0.f,0.f,0.f,0.f,0.f,0.f};
    v8f acc[8];
#pragma unroll
    for (int i = 0; i < 8; ++i) acc[i] = zero;

    for (int k0 = 0; k0 < C; k0 += KSTEP) {
        // Stage 128x32 f16 tiles of Q and K (512 16B chunks each, 2 per thread).
#pragma unroll
        for (int j = 0; j < 2; ++j) {
            const int ch = tid + j * 256;
            const int row = ch >> 2, sub = ch & 3;
            copy16_to_lds(Qs + row * LROW + sub * 8,
                          Qg + (size_t)row * C + k0 + sub * 8);
            copy16_to_lds(Ks + row * LROW + sub * 8,
                          Kg + (size_t)row * C + k0 + sub * 8);
            if (k0 + KSTEP < C) {
                __builtin_prefetch(Qg + (size_t)row * C + k0 + KSTEP + sub * 8, 0, 2);
                __builtin_prefetch(Kg + (size_t)row * C + k0 + KSTEP + sub * 8, 0, 2);
            }
        }
        staging_commit();
        __syncthreads();

        // B fragment: column m = w*16 + r, K chunks {8g..8g+7} and {16+8g..}.
        Frag bf;
        const _Float16* bp = Ks + (w * 16 + r) * LROW;
        bf.q[0] = *(const int4*)(bp + 8 * g);
        bf.q[1] = *(const int4*)(bp + 16 + 8 * g);

#pragma unroll
        for (int ni = 0; ni < 8; ++ni) {
            // A fragment: row n = ni*16 + r, K chunks {8g..8g+7} and {16+8g..}.
            Frag af;
            const _Float16* ap = Qs + (ni * 16 + r) * LROW;
            af.q[0] = *(const int4*)(ap + 8 * g);
            af.q[1] = *(const int4*)(ap + 16 + 8 * g);
            acc[ni] = __builtin_amdgcn_wmma_f32_16x16x32_f16(
                false, af.v, false, bf.v, (short)0, acc[ni], false, false);
        }
        __syncthreads();
    }

    // Epilogue: D VGPR i -> (row = ni*16 + i + 8g, col = w*16 + r). Mask fused.
    // Each half-wave writes a 64B contiguous run along m -> coalesced.
#pragma unroll
    for (int ni = 0; ni < 8; ++ni) {
#pragma unroll
        for (int i = 0; i < 8; ++i) {
            const int n = n0 + ni * 16 + i + 8 * g;
            const int m = m0 + w * 16 + r;
            float x = acc[ni][i];
            if (x == 0.0f) x = -100000.0f;
            sim[((size_t)b * N + n) * (size_t)M + m] = x;
        }
    }
}

// ---------------------------------------------------------------------------
// Kernel 3: in-place row softmax over M=4096.  One 256-thread block per row.
// ---------------------------------------------------------------------------
__global__ __launch_bounds__(256) void softmax_kernel(float* __restrict__ sim)
{
    const size_t row = blockIdx.x;
    float* p = sim + row * (size_t)M;
    const int tid = threadIdx.x;

    float4 v[4];
#pragma unroll
    for (int i = 0; i < 4; ++i)
        v[i] = ((const float4*)p)[i * 256 + tid];

    float mx = -3.4e38f;
#pragma unroll
    for (int i = 0; i < 4; ++i) {
        mx = fmaxf(mx, fmaxf(fmaxf(v[i].x, v[i].y), fmaxf(v[i].z, v[i].w)));
    }
#pragma unroll
    for (int off = 16; off > 0; off >>= 1)
        mx = fmaxf(mx, __shfl_xor(mx, off, 32));

    __shared__ float redmax[8];
    __shared__ float redsum[8];
    if ((tid & 31) == 0) redmax[tid >> 5] = mx;
    __syncthreads();
    mx = redmax[0];
#pragma unroll
    for (int j = 1; j < 8; ++j) mx = fmaxf(mx, redmax[j]);

    float s = 0.f;
#pragma unroll
    for (int i = 0; i < 4; ++i) {
        v[i].x = __expf(v[i].x - mx);
        v[i].y = __expf(v[i].y - mx);
        v[i].z = __expf(v[i].z - mx);
        v[i].w = __expf(v[i].w - mx);
        s += v[i].x + v[i].y + v[i].z + v[i].w;
    }
#pragma unroll
    for (int off = 16; off > 0; off >>= 1)
        s += __shfl_xor(s, off, 32);
    if ((tid & 31) == 0) redsum[tid >> 5] = s;
    __syncthreads();
    s = 0.f;
#pragma unroll
    for (int j = 0; j < 8; ++j) s += redsum[j];

    const float inv = 1.0f / s;
#pragma unroll
    for (int i = 0; i < 4; ++i) {
        v[i].x *= inv; v[i].y *= inv; v[i].z *= inv; v[i].w *= inv;
        ((float4*)p)[i * 256 + tid] = v[i];
    }
}

// ---------------------------------------------------------------------------
// Kernel 4: wg_t[b][c][n] = gamma * sum_m P[b][n][m] * V[b][c][m]
// A = P (fp32 -> f16 during LDS staging), B = V (f16, m-contiguous rows).
// Epilogue transposes each 128c x 16n D-subtile through LDS for coalesced
// stores along n.
// ---------------------------------------------------------------------------
__global__ __launch_bounds__(256) void pv_kernel(
    const float* __restrict__ P, const _Float16* __restrict__ V,
    const float* __restrict__ gamma, float* __restrict__ out)
{
    __shared__ _Float16 Ps[TILE * LROW];
    __shared__ _Float16 Vs[TILE * LROW];
    __shared__ float    Ds[TILE * 17];     // 128 c-rows x 16 n (pad 17: conflict-free)

    const int tid = threadIdx.x;
    const int b   = blockIdx.z;
    const int n0  = blockIdx.y * TILE;
    const int c0  = blockIdx.x * TILE;

    const float*    Pg = P + ((size_t)b * N + n0) * (size_t)M;
    const _Float16* Vg = V + ((size_t)b * C + c0) * (size_t)M;

    const int w = tid >> 5, lane = tid & 31, g = lane >> 4, r = lane & 15;

    const v8f zero = {0.f,0.f,0.f,0.f,0.f,0.f,0.f,0.f};
    v8f acc[8];
#pragma unroll
    for (int i = 0; i < 8; ++i) acc[i] = zero;

    for (int k0 = 0; k0 < M; k0 += KSTEP) {
        // Stage V tile via async DMA: 128 rows x 32 f16 (512 16B chunks, 2/thread).
#pragma unroll
        for (int j = 0; j < 2; ++j) {
            const int ch  = tid + j * 256;
            const int row = ch >> 2, sub = ch & 3;
            copy16_to_lds(Vs + row * LROW + sub * 8,
                          Vg + (size_t)row * M + k0 + sub * 8);
        }
        // Stage P tile (needs fp32->f16 convert, so VGPR path):
        // 128 rows x 32 fp32 (1024 float4 chunks, 4/thread).
#pragma unroll
        for (int j = 0; j < 4; ++j) {
            const int ch  = tid + j * 256;
            const int row = ch >> 3, sub = ch & 7;
            float4 f = *(const float4*)(Pg + (size_t)row * M + k0 + sub * 4);
            _Float16* dst = Ps + row * LROW + sub * 4;
            dst[0] = (_Float16)f.x; dst[1] = (_Float16)f.y;
            dst[2] = (_Float16)f.z; dst[3] = (_Float16)f.w;
        }
        staging_commit();
        __syncthreads();

        Frag bf;  // column c = w*16 + r, K(m) chunks {8g..8g+7} and {16+8g..}
        const _Float16* bp = Vs + (w * 16 + r) * LROW;
        bf.q[0] = *(const int4*)(bp + 8 * g);
        bf.q[1] = *(const int4*)(bp + 16 + 8 * g);

#pragma unroll
        for (int ni = 0; ni < 8; ++ni) {
            Frag af;  // row n = ni*16 + r
            const _Float16* ap = Ps + (ni * 16 + r) * LROW;
            af.q[0] = *(const int4*)(ap + 8 * g);
            af.q[1] = *(const int4*)(ap + 16 + 8 * g);
            acc[ni] = __builtin_amdgcn_wmma_f32_16x16x32_f16(
                false, af.v, false, bf.v, (short)0, acc[ni], false, false);
        }
        __syncthreads();
    }

    // Coalesced transposed epilogue with gamma fused.
    const float gm = gamma[0];
#pragma unroll
    for (int ni = 0; ni < 8; ++ni) {
        __syncthreads();
        // D element (n_local = ni*16 + i + 8g, c_local = w*16 + r) -> Ds[c][n16]
#pragma unroll
        for (int i = 0; i < 8; ++i)
            Ds[(w * 16 + r) * 17 + i + 8 * g] = gm * acc[ni][i];
        __syncthreads();
        // 128 rows x 16 floats: thread t writes 8 consecutive n of row c = t/2.
        const int c  = tid >> 1;
        const int h8 = (tid & 1) * 8;
        const float* s = Ds + c * 17 + h8;
        float4 o0 = { s[0], s[1], s[2], s[3] };
        float4 o1 = { s[4], s[5], s[6], s[7] };
        float* op = out + ((size_t)b * C + c0 + c) * (size_t)N + n0 + ni * 16 + h8;
        *(float4*)op       = o0;
        *(float4*)(op + 4) = o1;
    }
}

// ---------------------------------------------------------------------------
extern "C" void kernel_launch(void* const* d_in, const int* in_sizes, int n_in,
                              void* d_out, int out_size, void* d_ws, size_t ws_size,
                              hipStream_t stream)
{
    const float* xT    = (const float*)d_in[0];  // [B,C,N]
    const float* xR    = (const float*)d_in[1];  // [B,C,M]
    const float* xRGB  = (const float*)d_in[2];  // [B,C,M]
    const float* wq    = (const float*)d_in[3];
    const float* bq    = (const float*)d_in[4];
    const float* wk    = (const float*)d_in[5];
    const float* bk    = (const float*)d_in[6];
    const float* wv    = (const float*)d_in[7];
    const float* bv    = (const float*)d_in[8];
    const float* gamma = (const float*)d_in[9];

    float* wg_t = (float*)d_out;                       // [B,C,N]
    float* sim  = wg_t + (size_t)BB * C * N;           // [B,N,M] (raw -> softmaxed in place)

    // Workspace: Q [b][n][C], K [b][m][C], V [b][c][M]  (f16) = 12 MiB total
    _Float16* Qws = (_Float16*)d_ws;
    _Float16* Kws = Qws + (size_t)BB * N * C;
    _Float16* Vws = Kws + (size_t)BB * M * C;

    dim3 blk(256);

    dim3 pg(N / 64, C / 4, BB);
    proj_kernel<<<pg, blk, 0, stream>>>(xT,   wq, bq, Qws, 0);
    proj_kernel<<<pg, blk, 0, stream>>>(xR,   wk, bk, Kws, 0);
    proj_kernel<<<pg, blk, 0, stream>>>(xRGB, wv, bv, Vws, 1);

    dim3 sg(M / TILE, N / TILE, BB);          // 32 x 32 x 2
    sim_kernel<<<sg, blk, 0, stream>>>(Qws, Kws, sim);

    softmax_kernel<<<dim3(BB * N), blk, 0, stream>>>(sim);

    dim3 vg(C / TILE, N / TILE, BB);          // 2 x 32 x 2
    pv_kernel<<<vg, blk, 0, stream>>>(sim, Vws, gamma, wg_t);
}
